// ACMLoss_72636486909968
// MI455X (gfx1250) — matrix operand order repaired and checked by reading, not compiled, not run on hardware
//
#include <hip/hip_runtime.h>
#include <cstddef>

// ---------------------------------------------------------------------------
// ACM loss for MI455X (gfx1250, wave32).
//
//  * 96 cosine-sim GEMMs ([128,2048]x[2048,128]) with v_wmma_f32_16x16x32_bf16
//    (f32 -> bf16 on the fly), one 16x16 tile per wave, 8 waves per block,
//    sim matrix resident in LDS.
//  * 128x128 DP recurrences (LCS / soft-DP) fused in the same block using a
//    rolling 3-diagonal wavefront (255 anti-diagonal steps).
//  * Normalization uses precomputed RECIPROCAL row norms -> multiplies only,
//    no v_div_scale chains.
//  * FSD m+g sims = 128KB -> dynamic LDS (CDNA5: 320KB per WGP).
// ---------------------------------------------------------------------------

typedef __attribute__((ext_vector_type(16))) __bf16 v16bf;
typedef __attribute__((ext_vector_type(8)))  float  v8f;

#define LL   128
#define DD   2048
#define HD   1024
#define BB   32
#define TT   2048
#define NC1  21

// ws layout (float offsets) — all norm slots hold RECIPROCAL norms.
enum {
  WS_LCSN   = 0,        // 32*128 1/||row||, lcs_candi (full D)
  WS_FAN1   = 4096,     // fsd_act half-1 reciprocal norms
  WS_FAN2   = 8192,     // fsd_act half-2
  WS_FBN1   = 12288,    // fsd_bak half-1
  WS_FBN2   = 16384,    // fsd_bak half-2
  WS_GUIDE  = 20480,    // [32]
  WS_SPARSE = 20512,    // [32]
  WS_FEAT   = 20544,    // [32]
  WS_CLS    = 20576,    // [1]
  WS_LCSRES = 20608,    // [32]  (0..15 pos, 16..31 neg)
  WS_FSDRES = 20640,    // [32]  (0..15 act_act, 16..31 act_bak)
};

// ---------------------------------------------------------------------------
// WMMA tile accumulate: acc += A_tile(16 x K) * B_tile(16 x K)^T, bf16 inputs.
// A operand (16x32 bf16): lanes 0-15 row M=lane, K = 0-7,16-23;
// lanes 16-31 same rows, K = 8-15,24-31.  B operand (32x16): lanes hold
// column N=lane%16; lanes 0-15 K=0-7,16-23, lanes 16-31 K=8-15,24-31.
// ---------------------------------------------------------------------------
__device__ inline v8f wmma_accum(const float* __restrict__ Arow0,
                                 const float* __restrict__ Brow0,
                                 int kcount, v8f acc) {
  const int lane = threadIdx.x & 31;
  const int half = lane >> 4;
  const int lm   = lane & 15;
  const float* a = Arow0 + (size_t)lm * DD;
  const float* b = Brow0 + (size_t)lm * DD;
  #pragma unroll 2
  for (int kb = 0; kb < kcount; kb += 32) {
    v16bf af, bf;
    const float* ap = a + kb + half * 8;
    #pragma unroll
    for (int e = 0; e < 8; ++e) {
      af[e]     = (__bf16)ap[e];
      af[e + 8] = (__bf16)ap[16 + e];
    }
    const float* bp = b + kb + half * 8;
    #pragma unroll
    for (int e = 0; e < 8; ++e) {
      bf[e]     = (__bf16)bp[e];
      bf[e + 8] = (__bf16)bp[16 + e];
    }
    acc = __builtin_amdgcn_wmma_f32_16x16x32_bf16(
        /*neg_a=*/false, af, /*neg_b=*/false, bf,
        /*c_mod=*/(short)0, acc, /*reuse_a=*/false, /*reuse_b=*/false);
  }
  return acc;
}

// ---------------------------------------------------------------------------
// Reciprocal row norms: one block per (video, row).  5 reductions:
//   lcs full-D, fsd_act half1/half2, fsd_bak half1/half2.
// ---------------------------------------------------------------------------
__global__ void rownorm_kernel(const float* __restrict__ lcs,
                               const float* __restrict__ fa,
                               const float* __restrict__ fb,
                               float* __restrict__ ws) {
  const int id = blockIdx.x;            // 0..4095
  const size_t base = (size_t)id * DD;
  const float* lrow = lcs + base;
  const float* arow = fa + base;
  const float* brow = fb + base;
  float sl = 0.f, sa1 = 0.f, sa2 = 0.f, sb1 = 0.f, sb2 = 0.f;
  for (int i = threadIdx.x; i < DD; i += 256) {
    float v = lrow[i]; sl += v * v;
    float a = arow[i]; float bv = brow[i];
    if (i < HD) { sa1 += a * a; sb1 += bv * bv; }
    else        { sa2 += a * a; sb2 += bv * bv; }
  }
  __shared__ float red[5][256];
  red[0][threadIdx.x] = sl;  red[1][threadIdx.x] = sa1;
  red[2][threadIdx.x] = sa2; red[3][threadIdx.x] = sb1;
  red[4][threadIdx.x] = sb2;
  __syncthreads();
  for (int s = 128; s > 0; s >>= 1) {
    if (threadIdx.x < s) {
      #pragma unroll
      for (int k = 0; k < 5; ++k)
        red[k][threadIdx.x] += red[k][threadIdx.x + s];
    }
    __syncthreads();
  }
  if (threadIdx.x == 0) {
    ws[WS_LCSN + id] = rsqrtf(red[0][0]);
    ws[WS_FAN1 + id] = rsqrtf(red[1][0]);
    ws[WS_FAN2 + id] = rsqrtf(red[2][0]);
    ws[WS_FBN1 + id] = rsqrtf(red[3][0]);
    ws[WS_FBN2 + id] = rsqrtf(red[4][0]);
  }
}

// ---------------------------------------------------------------------------
// cls losses (tiny): one wave, one thread per batch row.
// ---------------------------------------------------------------------------
__global__ void cls_kernel(const float* __restrict__ xi,
                           const float* __restrict__ xc,
                           const float* __restrict__ xb,
                           const float* __restrict__ vl,
                           float* __restrict__ ws) {
  __shared__ float red[32];
  const int b = threadIdx.x;
  float vsum = 0.f;
  for (int c = 0; c < NC1 - 1; ++c) vsum += vl[b * (NC1 - 1) + c];
  const float inv_i = 1.f / vsum;
  const float inv_c = 1.f / (vsum + 1.f);
  float si = 0.f, sc = 0.f, sb = 0.f;
  for (int c = 0; c < NC1; ++c) {
    float v  = (c < NC1 - 1) ? vl[b * (NC1 - 1) + c] : 0.f;
    float li = v * inv_i;
    float lc = ((c < NC1 - 1) ? v : 1.f) * inv_c;
    float lb = (c < NC1 - 1) ? 0.f : 1.f;
    si += logf(xi[b * NC1 + c] + 1e-45f) * li;
    sc += logf(xc[b * NC1 + c] + 1e-45f) * lc;
    sb += logf(xb[b * NC1 + c] + 1e-45f) * lb;
  }
  red[b] = si + sc + sb;
  __syncthreads();
  if (b == 0) {
    float t = 0.f;
    for (int k = 0; k < 32; ++k) t += red[k];
    ws[WS_CLS] = -t / 32.f;   // sum of the three per-head means
  }
}

// ---------------------------------------------------------------------------
// guide + sparse sums, one block per batch row.
// ---------------------------------------------------------------------------
__global__ void guide_sparse_kernel(const float* __restrict__ cas,
                                    const float* __restrict__ att,
                                    float* __restrict__ ws) {
  const int b = blockIdx.x;
  float g = 0.f, sp = 0.f;
  for (int t = threadIdx.x; t < TT; t += 256) {
    float casv = cas[((size_t)b * TT + t) * NC1 + (NC1 - 1)];
    float a0   = att[((size_t)b * TT + t) * 3 + 0];
    float a1   = att[((size_t)b * TT + t) * 3 + 1];
    g  += fabsf(1.f - casv - a0);
    sp += a0 + a1;
  }
  __shared__ float r1[256], r2[256];
  r1[threadIdx.x] = g; r2[threadIdx.x] = sp;
  __syncthreads();
  for (int s = 128; s > 0; s >>= 1) {
    if (threadIdx.x < s) {
      r1[threadIdx.x] += r1[threadIdx.x + s];
      r2[threadIdx.x] += r2[threadIdx.x + s];
    }
    __syncthreads();
  }
  if (threadIdx.x == 0) { ws[WS_GUIDE + b] = r1[0]; ws[WS_SPARSE + b] = r2[0]; }
}

// ---------------------------------------------------------------------------
// feat loss per batch row.
// ---------------------------------------------------------------------------
__global__ void feat_kernel(const float* __restrict__ fi,
                            const float* __restrict__ fc,
                            const float* __restrict__ fb,
                            float* __restrict__ ws) {
  const int b = blockIdx.x;
  float si = 0.f, sc = 0.f, sb = 0.f;
  for (int i = threadIdx.x; i < DD; i += 256) {
    float a = fi[(size_t)b * DD + i]; si += a * a;
    float c = fc[(size_t)b * DD + i]; sc += c * c;
    float d = fb[(size_t)b * DD + i]; sb += d * d;
  }
  __shared__ float r1[256], r2[256], r3[256];
  r1[threadIdx.x] = si; r2[threadIdx.x] = sc; r3[threadIdx.x] = sb;
  __syncthreads();
  for (int s = 128; s > 0; s >>= 1) {
    if (threadIdx.x < s) {
      r1[threadIdx.x] += r1[threadIdx.x + s];
      r2[threadIdx.x] += r2[threadIdx.x + s];
      r3[threadIdx.x] += r3[threadIdx.x + s];
    }
    __syncthreads();
  }
  if (threadIdx.x == 0) {
    float ni = sqrtf(r1[0]), nc = sqrtf(r2[0]), nb = sqrtf(r3[0]);
    float f1 = fmaxf(50.f - ni + nc, 0.f);
    float f2 = fmaxf(50.f - nc + nb, 0.f);
    float v  = f1 + f2 + nb;
    ws[WS_FEAT + b] = v * v;
  }
}

// ---------------------------------------------------------------------------
// LCS pair kernel: WMMA sim into LDS, then anti-diagonal LCS DP.
// dynamic LDS = 128*128 sim (f32) + 3*132 diagonal floats = ~67 KB.
// ---------------------------------------------------------------------------
__global__ void lcs_pair_kernel(const float* __restrict__ cand,
                                const float* __restrict__ ws,
                                const int* __restrict__ pos,
                                const int* __restrict__ neg,
                                float* __restrict__ res) {
  extern __shared__ float smem[];
  float* sim  = smem;          // 16384
  float* diag = smem + 16384;  // 3*132
  const int p = blockIdx.x;
  const int* pr = (p < 16) ? (pos + 2 * p) : (neg + 2 * (p - 16));
  const int i0 = pr[0], i1 = pr[1];
  const float* A  = cand + (size_t)i0 * LL * DD;
  const float* Bm = cand + (size_t)i1 * LL * DD;
  const float* rna = ws + WS_LCSN + i0 * LL;   // reciprocal norms
  const float* rnb = ws + WS_LCSN + i1 * LL;

  const int wave = threadIdx.x >> 5;
  const int lane = threadIdx.x & 31;
  const int half = lane >> 4;
  const int lm   = lane & 15;

  for (int t = wave; t < 64; t += 8) {
    const int tm = t >> 3, tn = t & 7;
    v8f acc;
    #pragma unroll
    for (int r = 0; r < 8; ++r) acc[r] = 0.f;
    acc = wmma_accum(A + (size_t)tm * 16 * DD, Bm + (size_t)tn * 16 * DD, DD, acc);
    const int n = tn * 16 + lm;
    const float rnbv = rnb[n];
    #pragma unroll
    for (int r = 0; r < 8; ++r) {
      const int m = tm * 16 + half * 8 + r;
      float v = acc[r] * (rna[m] * rnbv);
      sim[m * LL + n] = fmaxf(0.f, v - 0.8f) * 5.f;   // pre-transformed s
    }
  }
  // zero diagonal buffers
  for (int i = threadIdx.x; i < 3 * 132; i += 256) diag[i] = 0.f;
  __syncthreads();

  float* p2 = diag;            // C on diagonal s-2
  float* p1 = diag + 132;      // s-1
  float* p0 = diag + 264;      // s
  const int tid = threadIdx.x;
  for (int s = 2; s <= 2 * LL; ++s) {
    if (tid < LL) {
      const int i = tid + 1;
      const int j = s - i;
      float nv = 0.f;
      if (j >= 1 && j <= LL) {
        const float c_diag = p2[i - 1];
        const float c_up   = p1[i - 1];
        const float c_left = p1[i];
        const float sij = sim[(i - 1) * LL + (j - 1)];
        nv = (sij > 0.5f) ? (c_diag + sij) : fmaxf(c_up, c_left);
        if (s == 2 * LL && i == LL) res[p] = nv;
      }
      p0[i] = nv;
    }
    __syncthreads();
    float* tmp = p2; p2 = p1; p1 = p0; p0 = tmp;
  }
}

// ---------------------------------------------------------------------------
// FSD pair kernel: two WMMA sims (m over first half, g over second half)
// into LDS, then soft-DP with logsumexp (gamma = 0.1).
// dynamic LDS = 2*16384 + 3*132 floats = ~132 KB (CDNA5: 320 KB per WGP).
// ---------------------------------------------------------------------------
__global__ void fsd_pair_kernel(const float* __restrict__ act,
                                const float* __restrict__ bak,
                                const float* __restrict__ ws,
                                const int* __restrict__ pos,
                                const int* __restrict__ neg,
                                float* __restrict__ res) {
  extern __shared__ float smem[];
  float* msim = smem;            // 16384
  float* gsim = smem + 16384;    // 16384
  float* diag = smem + 32768;    // 3*132
  const int p = blockIdx.x;
  int i0, i1;
  const float* Bbase;
  int nb1off, nb2off;
  if (p < 16) { i0 = pos[2 * p]; i1 = pos[2 * p + 1]; Bbase = act; nb1off = WS_FAN1; nb2off = WS_FAN2; }
  else        { int q = p - 16; i0 = neg[2 * q]; i1 = neg[2 * q + 1]; Bbase = bak; nb1off = WS_FBN1; nb2off = WS_FBN2; }
  const float* A    = act   + (size_t)i0 * LL * DD;
  const float* Bm   = Bbase + (size_t)i1 * LL * DD;
  const float* rna1 = ws + WS_FAN1 + i0 * LL;   // reciprocal norms
  const float* rna2 = ws + WS_FAN2 + i0 * LL;
  const float* rnb1 = ws + nb1off + i1 * LL;
  const float* rnb2 = ws + nb2off + i1 * LL;

  const int wave = threadIdx.x >> 5;
  const int lane = threadIdx.x & 31;
  const int half = lane >> 4;
  const int lm   = lane & 15;

  for (int t = wave; t < 64; t += 8) {
    const int tm = t >> 3, tn = t & 7;
    v8f accm, accg;
    #pragma unroll
    for (int r = 0; r < 8; ++r) { accm[r] = 0.f; accg[r] = 0.f; }
    accm = wmma_accum(A + (size_t)tm * 16 * DD,      Bm + (size_t)tn * 16 * DD,      HD, accm);
    accg = wmma_accum(A + (size_t)tm * 16 * DD + HD, Bm + (size_t)tn * 16 * DD + HD, HD, accg);
    const int n = tn * 16 + lm;
    const float rnb1v = rnb1[n], rnb2v = rnb2[n];
    #pragma unroll
    for (int r = 0; r < 8; ++r) {
      const int m = tm * 16 + half * 8 + r;
      msim[m * LL + n] = accm[r] * (rna1[m] * rnb1v);
      gsim[m * LL + n] = accg[r] * (rna2[m] * rnb2v);
    }
  }
  for (int i = threadIdx.x; i < 3 * 132; i += 256) diag[i] = 0.f;
  __syncthreads();

  const float G = 0.1f, invG = 10.f;
  float* p2 = diag;
  float* p1 = diag + 132;
  float* p0 = diag + 264;
  const int tid = threadIdx.x;
  for (int s = 2; s <= 2 * LL; ++s) {
    if (tid < LL) {
      const int i = tid + 1;
      const int j = s - i;
      float nv = 0.f;
      if (j >= 1 && j <= LL) {
        const float c_diag = p2[i - 1];
        const float c_up   = p1[i - 1];
        const float c_left = p1[i];
        const float m_ij = msim[(i - 1) * LL + (j - 1)];
        const float g_ij = gsim[(i - 1) * LL + (j - 1)];
        const float x0 = c_diag;
        const float x1 = g_ij + c_up;
        const float x2 = g_ij + c_left;
        const float mx = fmaxf(x0, fmaxf(x1, x2));
        const float ss = __expf((x0 - mx) * invG) + __expf((x1 - mx) * invG) +
                         __expf((x2 - mx) * invG);
        nv = m_ij + mx + G * __logf(ss);
        if (s == 2 * LL && i == LL) res[p] = nv;
      }
      p0[i] = nv;
    }
    __syncthreads();
    float* tmp = p2; p2 = p1; p1 = p0; p0 = tmp;
  }
}

// ---------------------------------------------------------------------------
// Final scalar assembly.
// ---------------------------------------------------------------------------
__global__ void finalize_kernel(const float* __restrict__ ws,
                                float* __restrict__ out) {
  if (threadIdx.x != 0 || blockIdx.x != 0) return;
  float guide = 0.f, sparse = 0.f, feat = 0.f;
  for (int b = 0; b < BB; ++b) {
    guide  += ws[WS_GUIDE + b];
    sparse += ws[WS_SPARSE + b];
    feat   += ws[WS_FEAT + b];
  }
  guide /= (float)BB; sparse /= (float)BB; feat /= (float)BB;
  const float cls = ws[WS_CLS];
  const float acm = cls + 1.0f * guide + 5e-5f * feat + 2e-4f * sparse;

  float pos_lcs = 0.f, neg_lcs = 0.f, act_act = 0.f, act_bak = 0.f;
  for (int k = 0; k < 16; ++k) {
    pos_lcs += ws[WS_LCSRES + k];
    neg_lcs += ws[WS_LCSRES + 16 + k];
    act_act += ws[WS_FSDRES + k];
    act_bak += ws[WS_FSDRES + 16 + k];
  }
  pos_lcs /= 16.f; neg_lcs /= 16.f; act_act /= 16.f; act_bak /= 16.f;
  out[0] = acm + 0.1f * (neg_lcs - pos_lcs) + 0.1f * (act_bak - act_act);
}

// ---------------------------------------------------------------------------
extern "C" void kernel_launch(void* const* d_in, const int* in_sizes, int n_in,
                              void* d_out, int out_size, void* d_ws, size_t ws_size,
                              hipStream_t stream) {
  (void)in_sizes; (void)n_in; (void)out_size; (void)ws_size;
  const float* act_inst_cls  = (const float*)d_in[0];
  const float* act_cont_cls  = (const float*)d_in[1];
  const float* act_back_cls  = (const float*)d_in[2];
  const float* vid_label     = (const float*)d_in[3];
  const float* temp_att      = (const float*)d_in[4];
  const float* act_inst_feat = (const float*)d_in[5];
  const float* act_cont_feat = (const float*)d_in[6];
  const float* act_back_feat = (const float*)d_in[7];
  const float* temp_cas      = (const float*)d_in[8];
  const float* lcs_candi     = (const float*)d_in[9];
  const float* fsd_act_candi = (const float*)d_in[10];
  const float* fsd_bak_candi = (const float*)d_in[11];
  const int*   pos_pair      = (const int*)d_in[12];
  const int*   neg_pair      = (const int*)d_in[13];
  float* ws  = (float*)d_ws;
  float* out = (float*)d_out;

  rownorm_kernel<<<BB * LL, 256, 0, stream>>>(lcs_candi, fsd_act_candi,
                                              fsd_bak_candi, ws);
  cls_kernel<<<1, 32, 0, stream>>>(act_inst_cls, act_cont_cls, act_back_cls,
                                   vid_label, ws);
  guide_sparse_kernel<<<BB, 256, 0, stream>>>(temp_cas, temp_att, ws);
  feat_kernel<<<BB, 256, 0, stream>>>(act_inst_feat, act_cont_feat,
                                      act_back_feat, ws);

  const size_t lcs_smem = (size_t)(16384 + 3 * 132) * sizeof(float);  // ~67 KB
  lcs_pair_kernel<<<32, 256, lcs_smem, stream>>>(lcs_candi, ws, pos_pair,
                                                 neg_pair, ws + WS_LCSRES);
  const size_t fsd_smem = (size_t)(32768 + 3 * 132) * sizeof(float);  // ~132 KB
  fsd_pair_kernel<<<32, 256, fsd_smem, stream>>>(fsd_act_candi, fsd_bak_candi,
                                                 ws, pos_pair, neg_pair,
                                                 ws + WS_FSDRES);
  finalize_kernel<<<1, 32, 0, stream>>>(ws, out);
}
